// SimpleDistanceLoss_69054484185809
// MI455X (gfx1250) — matrix operand order repaired and checked by reading, not compiled, not run on hardware
//
#include <hip/hip_runtime.h>
#include <hip/hip_bf16.h>

// ---------------------------------------------------------------------------
// SimpleDistanceLoss: sum_n min_m ||a_n - b_m||^2  via WMMA bf16x3 GEMM
//   N = M = 16384, D = 64, fp32 inputs, scalar fp32 output.
//   Inner loop tracks min_m (b^2 - 2ab); a^2 is added in the final reduce.
// ---------------------------------------------------------------------------

typedef __attribute__((ext_vector_type(16))) __bf16 v16bf;
typedef __attribute__((ext_vector_type(8)))  __bf16 v8bf;
typedef __attribute__((ext_vector_type(8)))  float  v8f;

#define DIM 64

// Order-preserving float->int key: signed-int compare == float compare.
__device__ inline int sdl_orderkey(float f)
{
    int i = __float_as_int(f);
    return (i >= 0) ? i : (int)(i ^ 0x7FFFFFFF);
}

// ---------------------------------------------------------------------------
// Kernel 1: split fp32 -> bf16 hi/lo, compute row norms, init row-min keys
// ---------------------------------------------------------------------------
__global__ void sdl_prep_kernel(const float* __restrict__ A, const float* __restrict__ B,
                                __bf16* __restrict__ ahi, __bf16* __restrict__ alo,
                                __bf16* __restrict__ bhi, __bf16* __restrict__ blo,
                                float* __restrict__ asq, float* __restrict__ bsq,
                                int* __restrict__ rowmin, int N, int M)
{
    int t = blockIdx.x * blockDim.x + threadIdx.x;
    if (t >= N + M) return;
    bool isB = (t >= N);
    int r = isB ? (t - N) : t;
    const float* src = (isB ? B : A) + (size_t)r * DIM;
    __bf16* hi = (isB ? bhi : ahi) + (size_t)r * DIM;
    __bf16* lo = (isB ? blo : alo) + (size_t)r * DIM;
    float s = 0.f;
#pragma unroll
    for (int d = 0; d < DIM; ++d) {
        float x = src[d];
        s = fmaf(x, x, s);
        __bf16 h = (__bf16)x;
        float hf = (float)h;
        __bf16 l = (__bf16)(x - hf);
        hi[d] = h;
        lo[d] = l;
    }
    if (isB) {
        bsq[r] = s;
    } else {
        asq[r] = s;
        rowmin[r] = 0x7F800000;  // key(+inf): larger than every finite key
    }
}

// ---------------------------------------------------------------------------
// A-fragment loader: 16-bit A 16x32, contiguous K order (matches B)
//   lanes 0-15 : elements 0..15 -> K=0..15
//   lanes 16-31: elements 0..15 -> K=16..31
// p already points at row*DIM + chunk*32 + (lane>>4)*16
// ---------------------------------------------------------------------------
__device__ inline v16bf sdl_load_afrag(const __bf16* p)
{
    v8bf a = *(const v8bf*)p;
    v8bf b = *(const v8bf*)(p + 8);
    v16bf r;
#pragma unroll
    for (int j = 0; j < 8; ++j) { r[j] = a[j]; r[j + 8] = b[j]; }
    return r;
}

// ---------------------------------------------------------------------------
// Kernel 2: tiled cross-term + row-min of (b^2 - 2ab).
//   256 threads = 8 waves; wave w handles TWO 16-row strips:
//     rows [blockIdx.x*256 + 32w, +32)  -> B fragments reused across strips
//   blockIdx.y splits the M columns into 8 ranges.
//   Per 16-column tile: 12x v_wmma_f32_16x16x32_bf16 (bf16x3, K=64, 2 strips)
//   against 8x global_load_b128 of B data.
// ---------------------------------------------------------------------------
__global__ __launch_bounds__(256) void sdl_distmin_kernel(
    const __bf16* __restrict__ ahi, const __bf16* __restrict__ alo,
    const __bf16* __restrict__ bhi, const __bf16* __restrict__ blo,
    const float* __restrict__ bsq,
    int* __restrict__ rowmin, int colsPerSplit)
{
    const int lane    = threadIdx.x & 31;
    const int wave    = threadIdx.x >> 5;
    const int halfsel = lane >> 4;   // 0: lanes 0-15, 1: lanes 16-31
    const int l15     = lane & 15;

    const int rowbase = blockIdx.x * 256 + wave * 32;   // strip0: +0, strip1: +16
    const int colbase = blockIdx.y * colsPerSplit;

    // --- A fragments for both strips (2 K-chunks x hi/lo x 2 strips) ---
    const int khalf = halfsel * 16;
    const __bf16* pah0 = ahi + (size_t)(rowbase + l15) * DIM + khalf;
    const __bf16* pal0 = alo + (size_t)(rowbase + l15) * DIM + khalf;
    const __bf16* pah1 = ahi + (size_t)(rowbase + 16 + l15) * DIM + khalf;
    const __bf16* pal1 = alo + (size_t)(rowbase + 16 + l15) * DIM + khalf;
    v16bf a0H0 = sdl_load_afrag(pah0);
    v16bf a0H1 = sdl_load_afrag(pah0 + 32);
    v16bf a0L0 = sdl_load_afrag(pal0);
    v16bf a0L1 = sdl_load_afrag(pal0 + 32);
    v16bf a1H0 = sdl_load_afrag(pah1);
    v16bf a1H1 = sdl_load_afrag(pah1 + 32);
    v16bf a1L0 = sdl_load_afrag(pal1);
    v16bf a1L1 = sdl_load_afrag(pal1 + 32);

    float vmin0[8], vmin1[8];
#pragma unroll
    for (int r = 0; r < 8; ++r) { vmin0[r] = 3.0e38f; vmin1[r] = 3.0e38f; }

    const int ntiles = colsPerSplit >> 4;
#pragma unroll 2
    for (int t = 0; t < ntiles; ++t) {
        // --- B fragments: 16-bit B 32x16 layout ---
        //   lane group 0 -> K=0..15 contiguous, group 1 -> K=16..31 contiguous
        const int col = colbase + t * 16 + l15;
        const __bf16* pbh = bhi + (size_t)col * DIM + halfsel * 16;
        const __bf16* pbl = blo + (size_t)col * DIM + halfsel * 16;
        v16bf bH0 = *(const v16bf*)pbh;          // K chunk 0
        v16bf bH1 = *(const v16bf*)(pbh + 32);   // K chunk 1
        v16bf bL0 = *(const v16bf*)pbl;
        v16bf bL1 = *(const v16bf*)(pbl + 32);
        float bs  = bsq[col];

        // cross = hi*hi + hi*lo + lo*hi, fp32 accum, K=64 in 2 chunks;
        // two accumulator chains (one per strip), interleaved for pipelining.
        v8f acc0 = {};
        v8f acc1 = {};
        acc0 = __builtin_amdgcn_wmma_f32_16x16x32_bf16(false, a0H0, false, bH0, (short)0, acc0, false, false);
        acc1 = __builtin_amdgcn_wmma_f32_16x16x32_bf16(false, a1H0, false, bH0, (short)0, acc1, false, false);
        acc0 = __builtin_amdgcn_wmma_f32_16x16x32_bf16(false, a0H1, false, bH1, (short)0, acc0, false, false);
        acc1 = __builtin_amdgcn_wmma_f32_16x16x32_bf16(false, a1H1, false, bH1, (short)0, acc1, false, false);
        acc0 = __builtin_amdgcn_wmma_f32_16x16x32_bf16(false, a0H0, false, bL0, (short)0, acc0, false, false);
        acc1 = __builtin_amdgcn_wmma_f32_16x16x32_bf16(false, a1H0, false, bL0, (short)0, acc1, false, false);
        acc0 = __builtin_amdgcn_wmma_f32_16x16x32_bf16(false, a0H1, false, bL1, (short)0, acc0, false, false);
        acc1 = __builtin_amdgcn_wmma_f32_16x16x32_bf16(false, a1H1, false, bL1, (short)0, acc1, false, false);
        acc0 = __builtin_amdgcn_wmma_f32_16x16x32_bf16(false, a0L0, false, bH0, (short)0, acc0, false, false);
        acc1 = __builtin_amdgcn_wmma_f32_16x16x32_bf16(false, a1L0, false, bH0, (short)0, acc1, false, false);
        acc0 = __builtin_amdgcn_wmma_f32_16x16x32_bf16(false, a0L1, false, bH1, (short)0, acc0, false, false);
        acc1 = __builtin_amdgcn_wmma_f32_16x16x32_bf16(false, a1L1, false, bH1, (short)0, acc1, false, false);

#pragma unroll
        for (int r = 0; r < 8; ++r) {
            float m0 = fmaf(-2.0f, acc0[r], bs);   // b^2 - 2ab (strip 0)
            float m1 = fmaf(-2.0f, acc1[r], bs);   // b^2 - 2ab (strip 1)
            vmin0[r] = fminf(vmin0[r], m0);
            vmin1[r] = fminf(vmin1[r], m1);
        }
    }

    // Row r (within a 16-strip) lives entirely in one 16-lane half:
    // min-reduce across those 16 lanes, then atomicMin (order-key) into
    // the global per-row minima.
#pragma unroll
    for (int r = 0; r < 8; ++r) {
        float v0 = vmin0[r];
        v0 = fminf(v0, __shfl_xor(v0, 1, 32));
        v0 = fminf(v0, __shfl_xor(v0, 2, 32));
        v0 = fminf(v0, __shfl_xor(v0, 4, 32));
        v0 = fminf(v0, __shfl_xor(v0, 8, 32));
        float v1 = vmin1[r];
        v1 = fminf(v1, __shfl_xor(v1, 1, 32));
        v1 = fminf(v1, __shfl_xor(v1, 2, 32));
        v1 = fminf(v1, __shfl_xor(v1, 4, 32));
        v1 = fminf(v1, __shfl_xor(v1, 8, 32));
        if (l15 == 0) {
            atomicMin(rowmin + rowbase +      halfsel * 8 + r, sdl_orderkey(v0));
            atomicMin(rowmin + rowbase + 16 + halfsel * 8 + r, sdl_orderkey(v1));
        }
    }
}

// ---------------------------------------------------------------------------
// Kernel 3: deterministic single-block sum of  asq[i] + min_m(b^2-2ab).
// ---------------------------------------------------------------------------
__global__ void sdl_reduce_kernel(const int* __restrict__ rowmin,
                                  const float* __restrict__ asq,
                                  float* __restrict__ out, int N)
{
    __shared__ float sm[256];
    float s = 0.f;
    for (int i = threadIdx.x; i < N; i += 256) {
        int k = rowmin[i];
        float v = __int_as_float((k >= 0) ? k : (int)(k ^ 0x7FFFFFFF));
        s += asq[i] + v;
    }
    sm[threadIdx.x] = s;
    __syncthreads();
    for (int k = 128; k > 0; k >>= 1) {
        if (threadIdx.x < k) sm[threadIdx.x] += sm[threadIdx.x + k];
        __syncthreads();
    }
    if (threadIdx.x == 0) out[0] = sm[0];
}

// ---------------------------------------------------------------------------
extern "C" void kernel_launch(void* const* d_in, const int* in_sizes, int n_in,
                              void* d_out, int out_size, void* d_ws, size_t ws_size,
                              hipStream_t stream)
{
    const float* A = (const float*)d_in[0];   // from_points [N, 64]
    const float* B = (const float*)d_in[1];   // to_points   [M, 64]
    const int N = in_sizes[0] / DIM;          // 16384
    const int M = in_sizes[1] / DIM;          // 16384

    // Workspace layout
    char* ws = (char*)d_ws;
    size_t off = 0;
    __bf16* ahi = (__bf16*)(ws + off); off += (size_t)N * DIM * sizeof(__bf16);
    __bf16* alo = (__bf16*)(ws + off); off += (size_t)N * DIM * sizeof(__bf16);
    __bf16* bhi = (__bf16*)(ws + off); off += (size_t)M * DIM * sizeof(__bf16);
    __bf16* blo = (__bf16*)(ws + off); off += (size_t)M * DIM * sizeof(__bf16);
    float*  asq = (float*)(ws + off);  off += (size_t)N * sizeof(float);
    float*  bsq = (float*)(ws + off);  off += (size_t)M * sizeof(float);
    int* rowmin = (int*)(ws + off);    off += (size_t)N * sizeof(int);

    // 1) bf16 hi/lo split + norms + row-min init
    {
        int total = N + M;
        sdl_prep_kernel<<<(total + 255) / 256, 256, 0, stream>>>(
            A, B, ahi, alo, bhi, blo, asq, bsq, rowmin, N, M);
    }

    // 2) WMMA cross tiles + per-row min of (b^2 - 2ab)
    {
        const int COLSPLITS = 8;
        dim3 grid(N / 256, COLSPLITS);   // 256 rows/block (8 waves x 2x16 rows)
        sdl_distmin_kernel<<<grid, 256, 0, stream>>>(
            ahi, alo, bhi, blo, bsq, rowmin, M / COLSPLITS);
    }

    // 3) sum of  a^2 + row minima  -> scalar
    sdl_reduce_kernel<<<1, 256, 0, stream>>>(rowmin, asq, (float*)d_out, N);
}
